// CausalSelfAttention_59184649338907
// MI455X (gfx1250) — matrix operand order repaired and checked by reading, not compiled, not run on hardware
//
#include <hip/hip_runtime.h>
#include <hip/hip_bf16.h>

// Causal self-attention for MI455X (gfx1250, wave32, WMMA + TDM).
// B=4, T=4096, D=H=1024. ~0.4 TFLOP => compute-bound on the matrix pipes.
// bf16 operands + v_wmma_f32_16x16x32_bf16 (8x deeper K than fp32 16x16x4),
// fp32 accumulate/softmax. Projection GEMM tiles are DMA'd into LDS by the
// Tensor Data Mover (double-buffered, hardware-padded rows), tracked with
// TENSORcnt.

#define BB 4
#define TT 4096
#define DD 1024
#define HH 1024
#define SCALE 0.03125f   // 1/sqrt(1024)

typedef __attribute__((ext_vector_type(16))) __bf16 v16bf;
typedef __attribute__((ext_vector_type(8)))  float  v8f;
typedef __attribute__((ext_vector_type(4)))  unsigned int u32x4;
typedef __attribute__((ext_vector_type(8)))  unsigned int u32x8;

union FragAB { v16bf v; uint4 q[2]; unsigned short u[16]; };
union Acc    { v8f v; float f[8]; };

__device__ __forceinline__ unsigned short f2bf(float f) {
  union { float f; unsigned u; } x; x.f = f;
  unsigned r = x.u + 0x7FFFu + ((x.u >> 16) & 1u);   // round-to-nearest-even
  return (unsigned short)(r >> 16);
}

__device__ __forceinline__ v8f wmma_bf16(const FragAB& a, const FragAB& b, v8f c) {
  return __builtin_amdgcn_wmma_f32_16x16x32_bf16(false, a.v, false, b.v,
                                                 (short)0, c, false, false);
}

// ---- Tensor Data Mover: 2D tile (rows x 32 bf16) global -> LDS ----
// Hardware pads +4 DWORDs after every 16 DWORDs (one 32-half row), giving the
// 40-half LDS row stride the WMMA fragment reads expect (bank-spread).
// D# layout per CDNA5 ISA ch.8: group0 {count, lds_addr, global_addr, type=2},
// group1 {data_size=2B, pad cfg, tensor dims, tile dims, row stride}.
__device__ __forceinline__ void tdm_load_2d(unsigned lds_off,
                                            const unsigned short* gaddr,
                                            unsigned tile_d1 /*rows*/,
                                            unsigned row_stride_elems) {
  unsigned long long ga = (unsigned long long)(size_t)gaddr;
  u32x4 g0;
  g0[0] = 1u;                                        // count=1, user descriptor
  g0[1] = lds_off;                                   // lds_addr (bytes)
  g0[2] = (unsigned)(ga & 0xffffffffu);              // global_addr[31:0]
  g0[3] = (unsigned)((ga >> 32) & 0x01ffffffu)       // global_addr[56:32]
        | 0x80000000u;                               // type = 2 ("image")
  const unsigned td0 = 0x7fffffffu, td1 = 0x7fffffffu;  // no OOB clipping
  u32x8 g1;
  g1[0] = (1u << 16)        // data_size = 2 bytes (bf16)
        | (1u << 20)        // pad_enable
        | (3u << 22)        // pad_interval: 16 DWORDs (one 64B row)
        | (3u << 25);       // pad_amount:   4 DWORDs (16B)
  g1[1] = (td0 & 0xffffu) << 16;                     // [47:32]=0, td0.lo
  g1[2] = (td0 >> 16) | ((td1 & 0xffffu) << 16);
  g1[3] = (td1 >> 16) | (32u << 16);                 // tile_dim0 = 32 elems
  g1[4] = tile_d1 & 0xffffu;                         // tile_dim1; tile_dim2=0
  g1[5] = row_stride_elems;                          // tensor_dim0_stride lo32
  g1[6] = 0u;                                        // stride hi16, dim1_stride lo
  g1[7] = 0u;
  asm volatile("tensor_load_to_lds %0, %1" :: "s"(g0), "s"(g1) : "memory");
}

// ---------------- fp32 -> bf16 (vectorized x4) ----------------
__global__ void k_cvt_bf16(const float* __restrict__ in,
                           unsigned short* __restrict__ out, int n4) {
  int i = blockIdx.x * blockDim.x + threadIdx.x;
  if (i >= n4) return;
  float4 f = ((const float4*)in)[i];
  ushort4 o;
  o.x = f2bf(f.x); o.y = f2bf(f.y); o.z = f2bf(f.z); o.w = f2bf(f.w);
  ((ushort4*)out)[i] = o;
}

// ------------- W (D x H) fp32 -> Wt (H x D) bf16 -------------
__global__ void k_cvt_w_t(const float* __restrict__ W,
                          unsigned short* __restrict__ Wt) {
  int idx = blockIdx.x * blockDim.x + threadIdx.x;   // over D*H
  int d = idx >> 10;
  int h = idx & (HH - 1);
  Wt[(size_t)h * DD + d] = f2bf(W[idx]);
}

// ---------------- projection GEMM: P = X * W ----------------
// X: (B*T) x D bf16 row-major.  Wt: H x D bf16.  Block = 128 thr (4 waves).
// Tile 128(M) x 64(N), K-step 32; wave w owns a 32x64 sub-block (8 WMMAs per
// step). Tiles are TDM-DMA'd into double-buffered padded LDS; wave 0 issues
// the descriptors (scalar branch; TDM ignores EXEC), completion flows through
// s_wait_tensorcnt + the workgroup barrier.
__global__ __launch_bounds__(128) void k_gemm_qkv(
    const unsigned short* __restrict__ X,
    const unsigned short* __restrict__ Wt,
    unsigned short* __restrict__ out,
    int transpose_out) {
  __shared__ __align__(16) unsigned short As[2][128 * 40];
  __shared__ __align__(16) unsigned short Bs[2][64 * 40];
  const int tid = threadIdx.x;
  const int w = tid >> 5, l = tid & 31;
  const int m0 = blockIdx.x * 128;
  const int n0 = blockIdx.y * 64;
  const bool is_w0 = (__builtin_amdgcn_readfirstlane(tid) >> 5) == 0;

  const unsigned short* gA = X + (size_t)m0 * DD;
  const unsigned short* gB = Wt + (size_t)n0 * DD;
  const unsigned ldsA[2] = { (unsigned)(size_t)&As[0][0],
                             (unsigned)(size_t)&As[1][0] };
  const unsigned ldsB[2] = { (unsigned)(size_t)&Bs[0][0],
                             (unsigned)(size_t)&Bs[1][0] };

  Acc acc[2][4];
#pragma unroll
  for (int m = 0; m < 2; ++m)
#pragma unroll
    for (int n = 0; n < 4; ++n)
#pragma unroll
      for (int r = 0; r < 8; ++r) acc[m][n].f[r] = 0.f;

  if (is_w0) {                       // prime buffer 0
    tdm_load_2d(ldsA[0], gA, 128, DD);
    tdm_load_2d(ldsB[0], gB, 64, DD);
  }

  for (int kk = 0; kk < DD; kk += 32) {
    const int buf = (kk >> 5) & 1;
    if (is_w0) __builtin_amdgcn_s_wait_tensorcnt(0);
    __syncthreads();                 // tile[buf] resident; tile[buf^1] consumed
    if (is_w0 && (kk + 32) < DD) {   // DMA next tile while computing this one
      tdm_load_2d(ldsA[buf ^ 1], gA + kk + 32, 128, DD);
      tdm_load_2d(ldsB[buf ^ 1], gB + kk + 32, 64, DD);
    }

    FragAB a[2];
#pragma unroll
    for (int m = 0; m < 2; ++m) {
      int row = w * 32 + m * 16 + (l & 15);
      int k0  = (l >> 4) * 8;                 // A: lane-half K offsets, +16
      a[m].q[0] = *(const uint4*)&As[buf][row * 40 + k0];
      a[m].q[1] = *(const uint4*)&As[buf][row * 40 + k0 + 16];
    }
#pragma unroll
    for (int n = 0; n < 4; ++n) {
      FragAB b;
      int nrow = n * 16 + (l & 15);
      int k0   = (l >> 4) * 8;                // B: lane-half K offsets, +16
      b.q[0] = *(const uint4*)&Bs[buf][nrow * 40 + k0];
      b.q[1] = *(const uint4*)&Bs[buf][nrow * 40 + k0 + 16];
#pragma unroll
      for (int m = 0; m < 2; ++m)
        acc[m][n].v = wmma_bf16(a[m], b, acc[m][n].v);
    }
  }

#pragma unroll
  for (int m = 0; m < 2; ++m) {
    const int mrow = m0 + w * 32 + m * 16 + ((l >> 4) * 8);
#pragma unroll
    for (int n = 0; n < 4; ++n) {
      int h = n0 + n * 16 + (l & 15);
      if (transpose_out) {
        int bb = mrow >> 12;
        int t  = mrow & (TT - 1);
        __align__(16) unsigned short tmp[8];
#pragma unroll
        for (int r = 0; r < 8; ++r) tmp[r] = f2bf(acc[m][n].f[r]);
        *(uint4*)(out + ((size_t)bb * HH + h) * TT + t) = *(const uint4*)tmp;
      } else {
#pragma unroll
        for (int r = 0; r < 8; ++r)
          out[(size_t)(mrow + r) * HH + h] = f2bf(acc[m][n].f[r]);
      }
    }
  }
}

// ---------------- scores + online softmax stats ----------------
__global__ __launch_bounds__(128) void k_attn_scores(
    const unsigned short* __restrict__ Q,    // (B,T,H) bf16
    const unsigned short* __restrict__ K,    // (B,T,H) bf16
    float* __restrict__ S,                   // (B,T,T) fp32
    float2* __restrict__ ml) {               // (B,T) {rowmax, rowsum}
  const int b = blockIdx.y;
  const int w = threadIdx.x >> 5, l = threadIdx.x & 31;
  const int q0 = blockIdx.x * 64 + w * 16;
  const unsigned short* Qb = Q + (size_t)b * TT * HH;
  const unsigned short* Kb = K + (size_t)b * TT * HH;
  float* Sb = S + (size_t)b * TT * TT;

  float m8[8], l8[8];
#pragma unroll
  for (int r = 0; r < 8; ++r) { m8[r] = -3.0e38f; l8[r] = 0.f; }

  const int qrow_a = q0 + (l & 15);
  const int qrow_c = q0 + ((l >> 4) * 8);

  for (int kt = 0; kt * 32 <= q0 + 15; ++kt) {
    const int key_a = kt * 32 + (l & 15);
    if ((kt + 1) * 32 <= q0 + 15)
      __builtin_prefetch(Kb + (size_t)(key_a + 32) * HH, 0, 1);

    Acc s0, s1;
#pragma unroll
    for (int r = 0; r < 8; ++r) { s0.f[r] = 0.f; s1.f[r] = 0.f; }

    for (int hh = 0; hh < HH; hh += 32) {
      FragAB a, b0, b1;
      const unsigned short* qa = Qb + (size_t)qrow_a * HH + hh + ((l >> 4) * 8);
      a.q[0] = *(const uint4*)qa;
      a.q[1] = *(const uint4*)(qa + 16);
      const unsigned short* kp0 = Kb + (size_t)key_a * HH + hh + ((l >> 4) * 8);
      b0.q[0] = *(const uint4*)kp0;
      b0.q[1] = *(const uint4*)(kp0 + 16);
      const unsigned short* kp1 = kp0 + (size_t)16 * HH;
      b1.q[0] = *(const uint4*)kp1;
      b1.q[1] = *(const uint4*)(kp1 + 16);
      s0.v = wmma_bf16(a, b0, s0.v);
      s1.v = wmma_bf16(a, b1, s1.v);
    }

    const int k0n = kt * 32 + (l & 15);
    const int k1n = k0n + 16;
#pragma unroll
    for (int r = 0; r < 8; ++r) {
      const int qr = qrow_c + r;
      float v0 = s0.f[r] * SCALE; if (k0n > qr) v0 = -1.0e30f;
      float v1 = s1.f[r] * SCALE; if (k1n > qr) v1 = -1.0e30f;
      Sb[(size_t)qr * TT + k0n] = v0;
      Sb[(size_t)qr * TT + k1n] = v1;
      float rm = fmaxf(v0, v1);
#pragma unroll
      for (int mk = 1; mk < 16; mk <<= 1) rm = fmaxf(rm, __shfl_xor(rm, mk, 32));
      float nm    = fmaxf(m8[r], rm);
      float alpha = __expf(m8[r] - nm);
      float p = __expf(v0 - nm) + __expf(v1 - nm);
#pragma unroll
      for (int mk = 1; mk < 16; mk <<= 1) p += __shfl_xor(p, mk, 32);
      l8[r] = l8[r] * alpha + p;
      m8[r] = nm;
    }
  }
  if ((l & 15) == 0) {
#pragma unroll
    for (int r = 0; r < 8; ++r)
      ml[(size_t)b * TT + qrow_c + r] = make_float2(m8[r], l8[r]);
  }
}

// ---------------- O = softmax(S) * V ----------------
__global__ __launch_bounds__(128) void k_attn_out(
    const float* __restrict__ S,
    const float2* __restrict__ ml,
    const unsigned short* __restrict__ Vt,   // (B,H,T) bf16
    float* __restrict__ O) {                 // (B,T,H) fp32
  const int b = blockIdx.z;
  const int w = threadIdx.x >> 5, l = threadIdx.x & 31;
  const int q0 = blockIdx.x * 64 + w * 16;
  const int h0 = blockIdx.y * 64;
  const float* Sb = S + (size_t)b * TT * TT;
  const unsigned short* Vb = Vt + (size_t)b * HH * TT;

  const int qa = q0 + (l & 15);
  float2 st = ml[(size_t)b * TT + qa];
  const float mrow = st.x;
  const float il   = 1.0f / st.y;

  Acc acc[4];
#pragma unroll
  for (int n = 0; n < 4; ++n)
#pragma unroll
    for (int r = 0; r < 8; ++r) acc[n].f[r] = 0.f;

  for (int kt = 0; kt * 32 <= q0 + 15; ++kt) {
    FragAB a;
    const float* sp = Sb + (size_t)qa * TT + kt * 32 + ((l >> 4) * 8);
    float4 sa0 = *(const float4*)sp;
    float4 sa1 = *(const float4*)(sp + 4);
    float4 sb0 = *(const float4*)(sp + 16);
    float4 sb1 = *(const float4*)(sp + 20);
    float fa[8] = {sa0.x, sa0.y, sa0.z, sa0.w, sa1.x, sa1.y, sa1.z, sa1.w};
    float fb[8] = {sb0.x, sb0.y, sb0.z, sb0.w, sb1.x, sb1.y, sb1.z, sb1.w};
#pragma unroll
    for (int r = 0; r < 8; ++r) {
      a.u[r]     = f2bf(__expf(fa[r] - mrow) * il);
      a.u[8 + r] = f2bf(__expf(fb[r] - mrow) * il);
    }
    const int kv = kt * 32 + ((l >> 4) * 8);
#pragma unroll
    for (int n = 0; n < 4; ++n) {
      FragAB bf;
      const unsigned short* vp = Vb + (size_t)(h0 + n * 16 + (l & 15)) * TT + kv;
      bf.q[0] = *(const uint4*)vp;
      bf.q[1] = *(const uint4*)(vp + 16);
      acc[n].v = wmma_bf16(a, bf, acc[n].v);
    }
  }

  const int qc = q0 + ((l >> 4) * 8);
#pragma unroll
  for (int n = 0; n < 4; ++n) {
    int h = h0 + n * 16 + (l & 15);
#pragma unroll
    for (int r = 0; r < 8; ++r)
      O[((size_t)b * TT + qc + r) * HH + h] = acc[n].f[r];
  }
}

extern "C" void kernel_launch(void* const* d_in, const int* in_sizes, int n_in,
                              void* d_out, int out_size, void* d_ws, size_t ws_size,
                              hipStream_t stream) {
  const float* x  = (const float*)d_in[0];
  const float* Wq = (const float*)d_in[1];
  const float* Wk = (const float*)d_in[2];
  const float* Wv = (const float*)d_in[3];
  float* out = (float*)d_out;

  char* ws = (char*)d_ws;
  size_t off = 0;
  unsigned short* xb  = (unsigned short*)(ws + off); off += (size_t)BB * TT * DD * 2;
  unsigned short* wqt = (unsigned short*)(ws + off); off += (size_t)DD * HH * 2;
  unsigned short* wkt = (unsigned short*)(ws + off); off += (size_t)DD * HH * 2;
  unsigned short* wvt = (unsigned short*)(ws + off); off += (size_t)DD * HH * 2;
  unsigned short* Qb  = (unsigned short*)(ws + off); off += (size_t)BB * TT * HH * 2;
  unsigned short* Kb  = (unsigned short*)(ws + off); off += (size_t)BB * TT * HH * 2;
  unsigned short* Vt  = (unsigned short*)(ws + off); off += (size_t)BB * TT * HH * 2;
  float*  S  = (float*)(ws + off);  off += (size_t)BB * TT * TT * 4;
  float2* ml = (float2*)(ws + off); off += (size_t)BB * TT * sizeof(float2);

  {
    int n4 = BB * TT * DD / 4;
    k_cvt_bf16<<<(n4 + 255) / 256, 256, 0, stream>>>(x, xb, n4);
  }
  k_cvt_w_t<<<(DD * HH) / 256, 256, 0, stream>>>(Wq, wqt);
  k_cvt_w_t<<<(DD * HH) / 256, 256, 0, stream>>>(Wk, wkt);
  k_cvt_w_t<<<(DD * HH) / 256, 256, 0, stream>>>(Wv, wvt);

  dim3 gg((BB * TT) / 128, HH / 64);
  k_gemm_qkv<<<gg, 128, 0, stream>>>(xb, wqt, Qb, 0);
  k_gemm_qkv<<<gg, 128, 0, stream>>>(xb, wkt, Kb, 0);
  k_gemm_qkv<<<gg, 128, 0, stream>>>(xb, wvt, Vt, 1);

  k_attn_scores<<<dim3(TT / 64, BB), 128, 0, stream>>>(Qb, Kb, S, ml);
  k_attn_out<<<dim3(TT / 64, HH / 64, BB), 128, 0, stream>>>(S, ml, Vt, out);
}
